// FeatureAggregation_4303557231293
// MI455X (gfx1250) — matrix operand order repaired and checked by reading, not compiled, not run on hardware
//
#include <hip/hip_runtime.h>
#include <hip/hip_bf16.h>
#include <math.h>

// ---------------------------------------------------------------------------
// FeatureAggregation for MI455X (gfx1250, wave32, WMMA).
// All 1x1 convs run as f16 WMMA GEMMs (v_wmma_f32_16x16x32_f16) with fp32
// accumulate; attention / LN / conv3x3 / upsample are memory-shaped VALU.
// GEMM uses 32-pixel N tiles with two independent accumulators per wave so
// consecutive v_wmma ops have no RAW dependency (XDL pipe stays busy).
// ---------------------------------------------------------------------------

typedef __attribute__((ext_vector_type(16))) _Float16 v16h;
typedef __attribute__((ext_vector_type(8)))  _Float16 v8h;
typedef __attribute__((ext_vector_type(8)))  float    v8f;

#define HH 96
#define WW 96
#define HW (HH*WW)          // 9216
#define NPIX (2*HW)         // 18432 pixels over both images
#define CF 128              // feature / hidden channels

__device__ __forceinline__ float gelu_exact(float v) {
    return 0.5f * v * (1.f + erff(v * 0.70710678118654752f));
}

// ---------------------------------------------------------------------------
// fp32 -> f16 conversion (weights)
// ---------------------------------------------------------------------------
__global__ __launch_bounds__(256) void f32_to_f16(const float* __restrict__ s,
                                                  _Float16* __restrict__ d, int n) {
    int i = blockIdx.x * 256 + threadIdx.x;
    if (i < n) d[i] = (_Float16)s[i];
}

// ---------------------------------------------------------------------------
// Direct conv3x3: in [2,256,24,24] -> out [2,122,24,24]
// grid = (122, 2), block = 256; weights for one output channel live in LDS.
// ---------------------------------------------------------------------------
__global__ __launch_bounds__(256) void conv3x3_k(const float* __restrict__ in,
                                                 const float* __restrict__ w,
                                                 const float* __restrict__ bias,
                                                 float* __restrict__ out) {
    const int o = blockIdx.x, b = blockIdx.y;
    __shared__ float sw[256 * 9];
    for (int i = threadIdx.x; i < 256 * 9; i += 256) sw[i] = w[(size_t)o * 2304 + i];
    __syncthreads();
    for (int p = threadIdx.x; p < 576; p += 256) {
        const int y = p / 24, x = p % 24;
        float acc = bias[o];
        for (int c = 0; c < 256; ++c) {
            const float* ib = in + ((size_t)(b * 256 + c)) * 576;
            const float* wr = &sw[c * 9];
            #pragma unroll
            for (int ky = 0; ky < 3; ++ky) {
                const int yy = y + ky - 1;
                if ((unsigned)yy < 24u) {
                    const float* r = ib + yy * 24;
                    #pragma unroll
                    for (int kx = 0; kx < 3; ++kx) {
                        const int xx = x + kx - 1;
                        if ((unsigned)xx < 24u) acc += wr[ky * 3 + kx] * r[xx];
                    }
                }
            }
        }
        out[((size_t)(b * 122 + o)) * 576 + p] = acc;
    }
}

// ---------------------------------------------------------------------------
// Bilinear x4 (align_corners=True) + GELU + concat(image, est_normals) -> nf
// nf layout [2,128,96,96]; grid = NPIX/256.
// ---------------------------------------------------------------------------
__global__ __launch_bounds__(256) void upsample_concat_k(const float* __restrict__ conv,
                                                         const float* __restrict__ img,
                                                         const float* __restrict__ nrm,
                                                         float* __restrict__ nf) {
    const int pg = blockIdx.x * 256 + threadIdx.x;
    const int b = pg / HW, p = pg % HW, y = p / WW, x = p % WW;
    const float r = 23.f / 95.f;
    const float fy = y * r; const int ly = (int)floorf(fy); const float ty = fy - ly;
    const int hy = min(ly + 1, 23);
    const float fx = x * r; const int lx = (int)floorf(fx); const float tx = fx - lx;
    const int hx = min(lx + 1, 23);
    const float w00 = (1.f - ty) * (1.f - tx), w01 = (1.f - ty) * tx;
    const float w10 = ty * (1.f - tx),         w11 = ty * tx;
    for (int c = 0; c < 122; ++c) {
        const float* cp = conv + ((size_t)(b * 122 + c)) * 576;
        float v = cp[ly * 24 + lx] * w00 + cp[ly * 24 + hx] * w01 +
                  cp[hy * 24 + lx] * w10 + cp[hy * 24 + hx] * w11;
        nf[((size_t)(b * CF + c)) * HW + p] = gelu_exact(v);
    }
    #pragma unroll
    for (int c = 0; c < 3; ++c) {
        nf[((size_t)(b * CF + 122 + c)) * HW + p] = img[((size_t)(b * 3 + c)) * HW + p];
        nf[((size_t)(b * CF + 125 + c)) * HW + p] = nrm[((size_t)(b * 3 + c)) * HW + p];
    }
}

// ---------------------------------------------------------------------------
// LayerNorm over channel dim (C=128): one wave per pixel, wave32 shfl reduce.
// grid = NPIX/8, block = 256 (8 waves).
// ---------------------------------------------------------------------------
__global__ __launch_bounds__(256) void layernorm_c_k(const float* __restrict__ X,
                                                     const float* __restrict__ gam,
                                                     const float* __restrict__ bet,
                                                     float* __restrict__ Y) {
    const int lane = threadIdx.x & 31;
    const int pg = blockIdx.x * 8 + (threadIdx.x >> 5);
    const int b = pg / HW, p = pg % HW;
    const float* xb = X + (size_t)b * CF * HW + p;
    float v[4];
    float s = 0.f;
    #pragma unroll
    for (int i = 0; i < 4; ++i) { v[i] = xb[(size_t)(lane + 32 * i) * HW]; s += v[i]; }
    #pragma unroll
    for (int m = 16; m >= 1; m >>= 1) s += __shfl_xor(s, m, 32);
    const float mean = s * (1.f / 128.f);
    float q = 0.f;
    #pragma unroll
    for (int i = 0; i < 4; ++i) { const float d = v[i] - mean; q += d * d; }
    #pragma unroll
    for (int m = 16; m >= 1; m >>= 1) q += __shfl_xor(q, m, 32);
    const float inv = rsqrtf(q * (1.f / 128.f) + 1e-5f);
    float* yb = Y + (size_t)b * CF * HW + p;
    #pragma unroll
    for (int i = 0; i < 4; ++i) {
        const int c = lane + 32 * i;
        yb[(size_t)c * HW] = (v[i] - mean) * inv * gam[c] + bet[c];
    }
}

// ---------------------------------------------------------------------------
// 1x1-conv GEMM via WMMA:  Y[b,m,p] = sum_k W[m,k] * X[b,k,p] + bias[m]
// Block = 256 threads (8 wave32 waves); grid.x = NPIX/32 (32-pixel N tiles),
// grid.y = M/128. Per block: stage X tile (fp32->f16) into LDS; each wave
// owns one 16-row M tile and TWO independent accumulators (two 16-pixel
// sub-tiles sharing one A fragment), so consecutive v_wmma are independent.
// Fragment index math follows CDNA5 ISA §7.12.2 layouts.
// EPI: 0 = bias, 1 = bias+GELU, 2 = bias+residual.
// ---------------------------------------------------------------------------
template <int EPI>
__global__ __launch_bounds__(256) void gemm1x1_wmma(const _Float16* __restrict__ Wh,
                                                    const float* __restrict__ bias,
                                                    const float* __restrict__ X,
                                                    const float* __restrict__ Res,
                                                    float* __restrict__ Y,
                                                    int M, int K) {
    __shared__ alignas(16) _Float16 sX[32][520];   // [pixel j][channel c], padded
    const int tid = threadIdx.x;
    const int tile = blockIdx.x;
    const int b  = (tile * 32) / HW;
    const int p0 = (tile * 32) % HW;

    // Stage activations (fp32 -> f16) for this 32-pixel tile.
    for (int idx = tid; idx < K * 32; idx += 256) {
        const int c = idx >> 5, j = idx & 31;
        sX[j][c] = (_Float16)X[((size_t)(b * K + c)) * HW + p0 + j];
    }
    __syncthreads();

    const int lane = tid & 31;
    const int g    = lane >> 4;     // lane half (ISA K-split)
    const int rn   = lane & 15;     // A row (M) / B col (N)
    const int m0   = (blockIdx.y * 8 + (tid >> 5)) * 16;

    v8f acc0 = {}, acc1 = {};
    const _Float16* aBase = Wh + (size_t)(m0 + rn) * K;
    #pragma unroll 2
    for (int k0 = 0; k0 < K; k0 += 32) {
        __builtin_prefetch(aBase + k0 + 64, 0, 3);   // global_prefetch on weight stream
        // A fragment: e<8 -> K = k0+8g+e ; e>=8 -> K = k0+16+8g+(e-8)
        const v8h alo = *(const v8h*)(aBase + k0 + 8 * g);
        const v8h ahi = *(const v8h*)(aBase + k0 + 16 + 8 * g);
        v16h A;
        #pragma unroll
        for (int e = 0; e < 8; ++e) { A[e] = alo[e]; A[8 + e] = ahi[e]; }
        // B fragments: e<8 -> K = k0+8g+e ; e>=8 -> K = k0+16+8g+(e-8)  (cols rn and rn+16)
        const _Float16* bp0 = &sX[rn][k0 + 8 * g];
        const _Float16* bp1 = &sX[16 + rn][k0 + 8 * g];
        const v8h b0lo = *(const v8h*)(bp0);
        const v8h b0hi = *(const v8h*)(bp0 + 16);
        const v8h b1lo = *(const v8h*)(bp1);
        const v8h b1hi = *(const v8h*)(bp1 + 16);
        v16h B0, B1;
        #pragma unroll
        for (int e = 0; e < 8; ++e) {
            B0[e] = b0lo[e]; B0[8 + e] = b0hi[e];
            B1[e] = b1lo[e]; B1[8 + e] = b1hi[e];
        }
        acc0 = __builtin_amdgcn_wmma_f32_16x16x32_f16(false, A, false, B0,
                                                      (short)0, acc0, false, false);
        acc1 = __builtin_amdgcn_wmma_f32_16x16x32_f16(false, A, false, B1,
                                                      (short)0, acc1, false, false);
    }

    // D layout: VGPR r -> M = 8g + r, N = lane&15.
    #pragma unroll
    for (int r = 0; r < 8; ++r) {
        const int m = m0 + 8 * g + r;
        const float bv = bias[m];
        const size_t row = ((size_t)(b * M + m)) * HW + p0 + rn;
        float y0 = acc0[r] + bv;
        float y1 = acc1[r] + bv;
        if (EPI == 1) { y0 = gelu_exact(y0); y1 = gelu_exact(y1); }
        if (EPI == 2) { y0 += Res[row]; y1 += Res[row + 16]; }
        Y[row]      = y0;
        Y[row + 16] = y1;
    }
}

// ---------------------------------------------------------------------------
// 5x5 local attention with per-pixel window mask.
// Zero-padded taps keep score 0 (matches unfold+zeropad); taps outside the
// Chebyshev window are -inf (excluded). One thread per pixel, coalesced
// channel-major loads. grid = NPIX/256.
// ---------------------------------------------------------------------------
__global__ __launch_bounds__(256) void local_attn5_k(const float* __restrict__ Q,
                                                     const float* __restrict__ Kf,
                                                     const float* __restrict__ Vf,
                                                     const int* __restrict__ wsz,
                                                     float* __restrict__ Out,
                                                     float scale) {
    const int pg = blockIdx.x * 256 + threadIdx.x;
    const int b = pg / HW, p = pg % HW, y = p / WW, x = p % WW;
    const int half = (wsz[pg] - 1) >> 1;

    unsigned inb = 0, act = 0;
    int off[25];
    #pragma unroll
    for (int t = 0; t < 25; ++t) {
        const int dy = t / 5 - 2, dx = t % 5 - 2;
        const bool ib = ((unsigned)(y + dy) < (unsigned)HH) &&
                        ((unsigned)(x + dx) < (unsigned)WW);
        off[t] = ib ? (dy * WW + dx) : 0;       // clamp so loads stay in-bounds
        if (ib) inb |= 1u << t;
        const int dist = max(abs(dy), abs(dx));
        if (dist <= half) act |= 1u << t;
    }

    float s[25];
    #pragma unroll
    for (int t = 0; t < 25; ++t) s[t] = 0.f;

    const float* qp = Q  + (size_t)b * CF * HW + p;
    const float* kp = Kf + (size_t)b * CF * HW + p;
    for (int c = 0; c < CF; ++c) {
        const float qv = qp[(size_t)c * HW];
        const float* kr = kp + (size_t)c * HW;
        #pragma unroll
        for (int t = 0; t < 25; ++t) {
            const float kv = kr[off[t]];
            s[t] += ((inb >> t) & 1u) ? qv * kv : 0.f;
        }
    }

    float mx = -3.4e38f;
    #pragma unroll
    for (int t = 0; t < 25; ++t) {
        s[t] *= scale;
        if ((act >> t) & 1u) mx = fmaxf(mx, s[t]);
    }
    float sum = 0.f;
    #pragma unroll
    for (int t = 0; t < 25; ++t) {
        s[t] = ((act >> t) & 1u) ? __expf(s[t] - mx) : 0.f;   // softmax weights
        sum += s[t];
    }
    const float rs = 1.f / sum;

    const unsigned use = act & inb;   // padded taps contribute zero value
    const float* vp = Vf + (size_t)b * CF * HW + p;
    float* op = Out + (size_t)b * CF * HW + p;
    for (int c = 0; c < CF; ++c) {
        const float* vr = vp + (size_t)c * HW;
        float a = 0.f;
        #pragma unroll
        for (int t = 0; t < 25; ++t) {
            const float vv = vr[off[t]];
            a += ((use >> t) & 1u) ? s[t] * vv : 0.f;
        }
        op[(size_t)c * HW] = a * rs;
    }
}

// ---------------------------------------------------------------------------
// Host-side orchestration
// ---------------------------------------------------------------------------
static inline void cvt(const float* s, _Float16* d, int n, hipStream_t st) {
    f32_to_f16<<<(n + 255) / 256, 256, 0, st>>>(s, d, n);
}
static inline void gemm(const _Float16* Wh, const float* bias, const float* X,
                        const float* Res, float* Y, int M, int K, int epi,
                        hipStream_t st) {
    dim3 grid(NPIX / 32, M / 128);
    if (epi == 0)      gemm1x1_wmma<0><<<grid, 256, 0, st>>>(Wh, bias, X, Res, Y, M, K);
    else if (epi == 1) gemm1x1_wmma<1><<<grid, 256, 0, st>>>(Wh, bias, X, Res, Y, M, K);
    else               gemm1x1_wmma<2><<<grid, 256, 0, st>>>(Wh, bias, X, Res, Y, M, K);
}

extern "C" void kernel_launch(void* const* d_in, const int* in_sizes, int n_in,
                              void* d_out, int out_size, void* d_ws, size_t ws_size,
                              hipStream_t stream) {
    // ---- inputs (setup_inputs dict order) ----
    const float* raw_in   = (const float*)d_in[0];   // [2,128,96,96]
    const float* nfeat    = (const float*)d_in[1];   // [2,256,24,24]
    const float* image    = (const float*)d_in[2];   // [2,3,96,96]
    const int*   wsizes   = (const int*)  d_in[3];   // [2,96,96]
    const float* estnrm   = (const float*)d_in[4];   // [2,3,96,96]
    const float* up_w     = (const float*)d_in[5];
    const float* up_b     = (const float*)d_in[6];
    const float* n1g = (const float*)d_in[7],  *n1b = (const float*)d_in[8];
    const float* q_w = (const float*)d_in[9],  *q_b = (const float*)d_in[10];
    const float* k_w = (const float*)d_in[11], *k_b = (const float*)d_in[12];
    const float* v_w = (const float*)d_in[13], *v_b = (const float*)d_in[14];
    const float* no_w = (const float*)d_in[15], *no_b = (const float*)d_in[16];
    const float* n2g = (const float*)d_in[17], *n2b = (const float*)d_in[18];
    const float* ip_w = (const float*)d_in[19], *ip_b = (const float*)d_in[20];
    const float* rk_w = (const float*)d_in[21], *rk_b = (const float*)d_in[22];
    const float* rv_w = (const float*)d_in[23], *rv_b = (const float*)d_in[24];
    const float* co_w = (const float*)d_in[25], *co_b = (const float*)d_in[26];
    const float* n3g = (const float*)d_in[27], *n3b = (const float*)d_in[28];
    const float* f1_w = (const float*)d_in[29], *f1_b = (const float*)d_in[30];
    const float* f2_w = (const float*)d_in[31], *f2_b = (const float*)d_in[32];
    const float* fin_w = (const float*)d_in[33], *fin_b = (const float*)d_in[34];
    float* out = (float*)d_out;

    // ---- workspace bump allocator ----
    char* wp = (char*)d_ws;
    auto alloc = [&](size_t bytes) -> char* {
        char* r = wp; wp += (bytes + 255) & ~(size_t)255; return r;
    };
    const size_t BIG = (size_t)CF * NPIX * sizeof(float);        // 9.44 MB
    _Float16* hq  = (_Float16*)alloc(128 * 128 * 2);
    _Float16* hk  = (_Float16*)alloc(128 * 128 * 2);
    _Float16* hv  = (_Float16*)alloc(128 * 128 * 2);
    _Float16* hno = (_Float16*)alloc(128 * 128 * 2);
    _Float16* hip_= (_Float16*)alloc(128 * 128 * 2);
    _Float16* hrk = (_Float16*)alloc(128 * 128 * 2);
    _Float16* hrv = (_Float16*)alloc(128 * 128 * 2);
    _Float16* hco = (_Float16*)alloc(128 * 128 * 2);
    _Float16* hfin= (_Float16*)alloc(128 * 128 * 2);
    _Float16* hf1 = (_Float16*)alloc(512 * 128 * 2);
    _Float16* hf2 = (_Float16*)alloc(128 * 512 * 2);
    float* convout = (float*)alloc((size_t)2 * 122 * 576 * sizeof(float));
    float* bA = (float*)alloc(BIG);   // nf / k2 / xn3 / ffn
    float* bB = (float*)alloc(BIG);   // xn / refined
    float* bC = (float*)alloc(BIG);   // q / inorm / v2
    float* bD = (float*)alloc(BIG);   // k / nnorm / cross
    float* bE = (float*)alloc(BIG);   // v / q2
    float* bF = (float*)alloc(BIG);   // attention outputs
    float* hbuf = (float*)alloc((size_t)512 * NPIX * sizeof(float));
    (void)ws_size; (void)n_in; (void)in_sizes; (void)out_size;

    const float scale = 0.08838834764831843f;   // 128^-0.5

    // ---- weights -> f16 ----
    cvt(q_w,  hq,  128*128, stream);  cvt(k_w,  hk,  128*128, stream);
    cvt(v_w,  hv,  128*128, stream);  cvt(no_w, hno, 128*128, stream);
    cvt(ip_w, hip_,128*128, stream);  cvt(rk_w, hrk, 128*128, stream);
    cvt(rv_w, hrv, 128*128, stream);  cvt(co_w, hco, 128*128, stream);
    cvt(fin_w,hfin,128*128, stream);  cvt(f1_w, hf1, 512*128, stream);
    cvt(f2_w, hf2, 128*512, stream);

    // ---- feature assembly ----
    conv3x3_k<<<dim3(122, 2), 256, 0, stream>>>(nfeat, up_w, up_b, convout);
    upsample_concat_k<<<NPIX / 256, 256, 0, stream>>>(convout, image, estnrm, bA); // nf

    // ---- self attention on normal features ----
    layernorm_c_k<<<NPIX / 8, 256, 0, stream>>>(bA, n1g, n1b, bB);        // xn
    gemm(hq, q_b, bB, nullptr, bC, 128, 128, 0, stream);                  // q
    gemm(hk, k_b, bB, nullptr, bD, 128, 128, 0, stream);                  // k
    gemm(hv, v_b, bB, nullptr, bE, 128, 128, 0, stream);                  // v
    local_attn5_k<<<NPIX / 256, 256, 0, stream>>>(bC, bD, bE, wsizes, bF, scale);
    gemm(hno, no_b, bF, bA, bB, 128, 128, 2, stream);                     // refined = proj + nf

    // ---- cross attention (image queries, refined keys/values) ----
    layernorm_c_k<<<NPIX / 8, 256, 0, stream>>>(raw_in, n2g, n2b, bC);    // inorm
    layernorm_c_k<<<NPIX / 8, 256, 0, stream>>>(bB, n2g, n2b, bD);        // nnorm
    gemm(hip_, ip_b, bC, nullptr, bE, 128, 128, 0, stream);               // q2
    gemm(hrk,  rk_b, bD, nullptr, bA, 128, 128, 0, stream);               // k2
    gemm(hrv,  rv_b, bD, nullptr, bC, 128, 128, 0, stream);               // v2
    local_attn5_k<<<NPIX / 256, 256, 0, stream>>>(bE, bA, bC, wsizes, bF, scale);
    gemm(hco, co_b, bF, raw_in, bD, 128, 128, 2, stream);                 // cross

    // ---- FFN ----
    layernorm_c_k<<<NPIX / 8, 256, 0, stream>>>(bD, n3g, n3b, bA);        // xn3
    gemm(hf1, f1_b, bA, nullptr, hbuf, 512, 128, 1, stream);              // gelu(f1)
    gemm(hf2, f2_b, hbuf, bD, bA, 128, 512, 2, stream);                   // ffn = f2 + cross

    // ---- final projection ----
    gemm(hfin, fin_b, bA, nullptr, out, 128, 128, 0, stream);
}
